// EdgeNetwork_49349174231510
// MI455X (gfx1250) — compile-verified, matrix operand order checked by reading, not run on hardware
//
#include <hip/hip_runtime.h>
#include <stdint.h>

typedef _Float16 v16h __attribute__((ext_vector_type(16)));
typedef __fp16   f16x2 __attribute__((ext_vector_type(2)));   // cvt_pkrtz result type
typedef float    v8f  __attribute__((ext_vector_type(8)));
typedef unsigned int v8u __attribute__((ext_vector_type(8)));

#define NF 16      // node feature size
#define NH 32      // node hidden size
#define IN_DIM 97  // 2*(16+32)+1
#define MID 80
#define OUTD 64
#define WAVES 8
#define TILE (WAVES * 16)

#define W1B_DW (5 * 4 * 32 * 8)          // 5120 dwords
#define W2B_DW (4 * 3 * 32 * 8)          // 3072 dwords
#define WB_DW  (W1B_DW + W2B_DW)         // 8192 dwords = 32 KB

__device__ __forceinline__ float fast_tanh(float x) {
#if __has_builtin(__builtin_amdgcn_tanhf)
  return __builtin_amdgcn_tanhf(x);
#else
  return tanhf(x);
#endif
}

__device__ __forceinline__ unsigned int pack2(float a, float b) {
  union { f16x2 h; unsigned int u; } c;
  c.h = __builtin_amdgcn_cvt_pkrtz(a, b);
  return c.u;
}

// Convert 16 f32 -> v16h fragment (pairs packed per dword)
__device__ __forceinline__ v16h cvt16(const float* f) {
  union { v16h v; f16x2 p[8]; } u;
#pragma unroll
  for (int i = 0; i < 8; ++i)
    u.p[i] = __builtin_amdgcn_cvt_pkrtz(f[2 * i], f[2 * i + 1]);
  return u.v;
}

// Load 8 consecutive concat-columns [c, c+8) of row r (c multiple of 8, c < 96)
__device__ __forceinline__ void load8(float* o, int c, long long r,
                                      const float* __restrict__ df,
                                      const float* __restrict__ dh,
                                      const float* __restrict__ sf,
                                      const float* __restrict__ sh) {
  const float* p;
  if (c < 16)      p = df + r * NF + c;
  else if (c < 48) p = dh + r * NH + (c - 16);
  else if (c < 64) p = sf + r * NF + (c - 48);
  else             p = sh + r * NH + (c - 64);
  float4 q0 = *(const float4*)p;
  float4 q1 = *(const float4*)(p + 4);
  o[0] = q0.x; o[1] = q0.y; o[2] = q0.z; o[3] = q0.w;
  o[4] = q1.x; o[5] = q1.y; o[6] = q1.z; o[7] = q1.w;
}

// ---------------------------------------------------------------------------
// Setup kernel: pack W1/W2 into WMMA B-fragment layout once, into d_ws.
// wpk[0..5119]  = W1 fragments (K 97->128, 5 N-tiles x 4 K-steps)
// wpk[5120..8191] = W2 fragments (K 80->96, 4 N-tiles x 3 K-steps)
// Fragment index = ((nt*KSTEPS + kk)*32 + lane)*8 + v ; per (lane) 8 dwords
// hold K pattern k = kk*32 + (v>>2)*16 + (lane>>4)*8 + (v&3)*2, n = nt*16+lane%16
// ---------------------------------------------------------------------------
__global__ __launch_bounds__(256) void pack_weights(
    const float* __restrict__ W1, const float* __restrict__ W2,
    unsigned int* __restrict__ wpk) {
  int tid = blockIdx.x * 256 + threadIdx.x;
  if (tid < W1B_DW) {
    int i = tid;
    int v = i & 7, ln = (i >> 3) & 31, ks = (i >> 8) & 3, nt = i >> 10;
    int n = nt * 16 + (ln & 15);
    int k = ks * 32 + (v >> 2) * 16 + (ln >> 4) * 8 + (v & 3) * 2;
    float f0 = (k     < IN_DIM) ? W1[n * IN_DIM + k]     : 0.f;
    float f1 = (k + 1 < IN_DIM) ? W1[n * IN_DIM + k + 1] : 0.f;
    wpk[tid] = pack2(f0, f1);
  } else if (tid < WB_DW) {
    int i = tid - W1B_DW;
    int v = i & 7, ln = (i >> 3) & 31, g = i >> 8;
    int nt = g / 3, ks = g % 3;
    int n = nt * 16 + (ln & 15);
    int k = ks * 32 + (v >> 2) * 16 + (ln >> 4) * 8 + (v & 3) * 2;
    float f0 = (k     < MID) ? W2[n * MID + k]     : 0.f;
    float f1 = (k + 1 < MID) ? W2[n * MID + k + 1] : 0.f;
    wpk[tid] = pack2(f0, f1);
  }
}

// ---------------------------------------------------------------------------
// Main kernel: 8 waves/block, each wave computes a 16-edge tile with
// v_wmma_f32_16x16x32_f16. Weights streamed ws -> LDS via async-to-LDS b128.
// ---------------------------------------------------------------------------
__global__ __launch_bounds__(256) void edge_mlp_wmma(
    const float* __restrict__ df, const float* __restrict__ dh,
    const float* __restrict__ sf, const float* __restrict__ sh,
    const float* __restrict__ ef,
    const unsigned int* __restrict__ wpk,
    const float* __restrict__ b1, const float* __restrict__ b2,
    float* __restrict__ out, int E) {
  __shared__ __align__(32) unsigned int WB[WB_DW];             // 32 KB fragments
  __shared__ float b1s[MID];
  __shared__ float b2s[OUTD];
  __shared__ __align__(16) unsigned int hbuf[WAVES][16 * 48];  // 16 rows x 96 f16

  const int t    = threadIdx.x;
  const int lane = t & 31;
  const int w    = t >> 5;
  const int hi   = lane >> 4;   // K-half selector
  const int ml   = lane & 15;   // A-row (m) / B-col (n) within tile

  // ---- Async-copy pre-packed weight fragments (L2-resident) into LDS.
  // 2048 uint4 chunks / 256 threads = 8 async b128 transfers each.
#pragma unroll
  for (int j = 0; j < 8; ++j) {
    int idx4 = t + j * 256;  // uint4 index, < 2048
    unsigned lds_addr = (unsigned)(size_t)&WB[idx4 * 4];
    unsigned long long gaddr = (unsigned long long)(size_t)(wpk + idx4 * 4);
    asm volatile("global_load_async_to_lds_b128 %0, %1, off"
                 :: "v"(lds_addr), "v"(gaddr)
                 : "memory");
  }
  if (t < MID)  b1s[t] = b1[t];
  if (t < OUTD) b2s[t] = b2[t];
  // zero-pad h columns [80,96) once (never overwritten; n < 80 in h stores)
#pragma unroll
  for (int j = 0; j < 4; ++j) {
    int idx = lane * 4 + j;  // 0..127
    hbuf[w][(idx >> 3) * 48 + 40 + (idx & 7)] = 0u;
  }

  // ---- Stage x A-fragments directly from global (f32 -> f16 packed)
  const long long base = (long long)blockIdx.x * TILE + w * 16;
  long long row = base + ml;
  if (row >= E) row = E - 1;  // clamp; out-of-range rows never stored

  float f[16];
  v16h a[4];
#pragma unroll
  for (int kk = 0; kk < 3; ++kk) {
    int c1 = kk * 32 + hi * 8;       // first 8-run
    load8(f,     c1,      row, df, dh, sf, sh);
    load8(f + 8, c1 + 16, row, df, dh, sf, sh);
    a[kk] = cvt16(f);
  }
  {  // kk == 3 covers cols 96..127: edge_feat at col 96, zeros beyond
    union { v16h v; unsigned int u[8]; } ua;
    ua.u[0] = (hi == 0) ? pack2(ef[row], 0.f) : 0u;
#pragma unroll
    for (int v = 1; v < 8; ++v) ua.u[v] = 0u;
    a[3] = ua.v;
  }

  // Wait for this wave's async weight transfers, then block barrier so every
  // wave sees the full fragment table + h pad.
  asm volatile("s_wait_asynccnt 0" ::: "memory");
  __syncthreads();

  // ---- Layer 1: h = relu(x W1^T + b1), 5 N-tiles x 4 K-steps
  _Float16* hp = (_Float16*)hbuf[w];
#pragma unroll
  for (int nt = 0; nt < 5; ++nt) {
    v8f c;
    float bias = b1s[nt * 16 + ml];
#pragma unroll
    for (int r = 0; r < 8; ++r) c[r] = bias;
#pragma unroll
    for (int kk = 0; kk < 4; ++kk) {
      union { v16h v; v8u u; } b;
      b.u = *(const v8u*)&WB[((nt * 4 + kk) * 32 + lane) * 8];
      c = __builtin_amdgcn_wmma_f32_16x16x32_f16(
          false, a[kk], false, b.v, (short)0, c, false, false);
    }
    // ReLU + scatter D (m per VGPR, n per lane) to LDS h buffer
#pragma unroll
    for (int r = 0; r < 8; ++r) {
      float hv = c[r] > 0.f ? c[r] : 0.f;
      int m = r + hi * 8;
      hp[m * 96 + nt * 16 + ml] = (_Float16)hv;
    }
  }

  __syncthreads();  // h tile complete (cross-lane transpose via LDS)

  // ---- Gather layer-2 A fragments (m per lane, k per VGPR) from h buffer
  const unsigned int* h32 = (const unsigned int*)hbuf[w];
  v16h a2[3];
#pragma unroll
  for (int kk = 0; kk < 3; ++kk) {
    union { v16h v; uint4 q[2]; } u2;
    int d = ml * 48 + kk * 16 + hi * 4;
    u2.q[0] = *(const uint4*)&h32[d];
    u2.q[1] = *(const uint4*)&h32[d + 8];
    a2[kk] = u2.v;
  }

  // ---- Layer 2: out = tanh(h W2^T + b2), 4 N-tiles x 3 K-steps
#pragma unroll
  for (int nt = 0; nt < 4; ++nt) {
    v8f c;
    float bias = b2s[nt * 16 + ml];
#pragma unroll
    for (int r = 0; r < 8; ++r) c[r] = bias;
#pragma unroll
    for (int kk = 0; kk < 3; ++kk) {
      union { v16h v; v8u u; } b;
      b.u = *(const v8u*)&WB[W1B_DW + ((nt * 3 + kk) * 32 + lane) * 8];
      c = __builtin_amdgcn_wmma_f32_16x16x32_f16(
          false, a2[kk], false, b.v, (short)0, c, false, false);
    }
#pragma unroll
    for (int r = 0; r < 8; ++r) {
      long long m = base + r + hi * 8;
      if (m < E) out[m * OUTD + nt * 16 + ml] = fast_tanh(c[r]);
    }
  }
}

extern "C" void kernel_launch(void* const* d_in, const int* in_sizes, int n_in,
                              void* d_out, int out_size, void* d_ws, size_t ws_size,
                              hipStream_t stream) {
  (void)n_in; (void)out_size; (void)ws_size;
  const float* df = (const float*)d_in[0];
  const float* dh = (const float*)d_in[1];
  const float* sf = (const float*)d_in[2];
  const float* sh = (const float*)d_in[3];
  const float* ef = (const float*)d_in[4];
  const float* W1 = (const float*)d_in[5];
  const float* b1 = (const float*)d_in[6];
  const float* W2 = (const float*)d_in[7];
  const float* b2 = (const float*)d_in[8];
  float* out = (float*)d_out;
  unsigned int* wpk = (unsigned int*)d_ws;  // 32 KB packed weight fragments
  int E = in_sizes[4];  // edge_feat flat count == E

  pack_weights<<<(WB_DW + 255) / 256, 256, 0, stream>>>(W1, W2, wpk);

  int grid = (E + TILE - 1) / TILE;
  edge_mlp_wmma<<<grid, 256, 0, stream>>>(df, dh, sf, sh, ef, wpk, b1, b2,
                                          out, E);
}
